// LayerProperties_55276229099580
// MI455X (gfx1250) — matrix operand order repaired and checked by reading, not compiled, not run on hardware
//
#include <hip/hip_runtime.h>
#include <hip/hip_bf16.h>

typedef __attribute__((ext_vector_type(16))) __bf16 v16bf;
typedef __attribute__((ext_vector_type(8)))  float  v8f;
typedef __attribute__((ext_vector_type(2)))  float  v2f;

#define NCH 29
#define HID 64
#define NBLK_PER_CH 64
#define WAVES 8

static __device__ __forceinline__ unsigned short f2bf(float x) {
    __bf16 b = (__bf16)x;                    // hardware f32->bf16 convert
    return __builtin_bit_cast(unsigned short, b);
}
static __device__ __forceinline__ float eluf(float x) {
    return x > 0.f ? x : (__expf(x) - 1.f);
}
static __device__ __forceinline__ float softplusf(float x) {
    return fmaxf(x, 0.f) + __logf(1.f + __expf(-fabsf(x)));
}

union BU { uint4 u[2]; v16bf v; };

__global__ __launch_bounds__(256) void lp_kernel(
    const float* __restrict__ tg_, const float* __restrict__ tl_, const float* __restrict__ ti_,
    const float* __restrict__ mu_, const float* __restrict__ mub_,
    const float* __restrict__ W1, const float* __restrict__ b1,
    const float* __restrict__ W2, const float* __restrict__ b2,
    const float* __restrict__ W3, const float* __restrict__ b3,
    float* __restrict__ out, int tilesPerWave)
{
    __shared__ float  sW1[256];          // [4][64] f32 (staging for B1 reg tiles)
    __shared__ float  sb1[64];
    __shared__ float  sb2[64];
    __shared__ float  sb3[16];
    __shared__ __bf16 sW2[4096];         // 256 B-slots * 16 bf16 (WMMA B layout)
    __shared__ __bf16 sW3[1024];         // 64 B-slots * 16 bf16
    __shared__ __bf16 sH[WAVES][16*64];  // per-wave activation tile
    __shared__ float  sLg[WAVES][48];    // per-wave logits, column-major [3][16]

    const int k    = blockIdx.x / NBLK_PER_CH;
    const int blk  = blockIdx.x % NBLK_PER_CH;
    const int t    = threadIdx.x;
    const int wave = t >> 5;
    const int lane = t & 31;
    const int m    = lane & 15;          // matrix row / column owned by this lane
    const int hiH  = (lane & 16) ? 1 : 0;

    // ---------------- per-block weight staging ----------------
    sW1[t] = W1[k*256 + t];
    if (t < 64) { sb1[t] = b1[k*64+t]; sb2[t] = b2[k*64+t]; }
    if (t < 16) sb3[t] = (t < 3) ? b3[k*3+t] : 0.f;
    {   // W2 -> bf16, pre-swizzled into WMMA B lane layout. slot = kt*128+nt*32+L
        const int kt = t >> 7, nt = (t >> 5) & 3, L = t & 31;
        const int colN  = nt*16 + (L & 15);
        const int kbase = kt*32 + ((L & 16) ? 16 : 0);
        const float* wsrc = W2 + (size_t)k*4096 + (size_t)kbase*64 + colN;
        unsigned short* dst = (unsigned short*)&sW2[t*16];
        #pragma unroll
        for (int j = 0; j < 16; ++j) dst[j] = f2bf(wsrc[j*64]);
    }
    if (t < 64) {   // W3 -> bf16 B layout, N padded 3->16 with zeros. slot = kt*32+L
        const int kt = t >> 5, L = t & 31;
        const int colN  = L & 15;
        const int kbase = kt*32 + ((L & 16) ? 16 : 0);
        unsigned short* dst = (unsigned short*)&sW3[t*16];
        #pragma unroll
        for (int j = 0; j < 16; ++j) {
            float v = (colN < 3) ? W3[(size_t)k*192 + (size_t)(kbase+j)*3 + colN] : 0.f;
            dst[j] = f2bf(v);
        }
    }
    __syncthreads();

    // B tiles resident in VGPRs for the whole sample loop
    v16bf B2t[2][4], B3t[2];
    #pragma unroll
    for (int kt = 0; kt < 2; ++kt) {
        #pragma unroll
        for (int nt = 0; nt < 4; ++nt) {
            const uint4* p = (const uint4*)&sW2[((kt*4+nt)*32 + lane)*16];
            BU bu; bu.u[0] = p[0]; bu.u[1] = p[1];
            B2t[kt][nt] = bu.v;
        }
        const uint4* q = (const uint4*)&sW3[(kt*32 + lane)*16];
        BU bu; bu.u[0] = q[0]; bu.u[1] = q[1];
        B3t[kt] = bu.v;
    }
    // W1 B-tiles (4x16 f32 each): lanes 0-15 hold K=0,1; lanes 16-31 hold K=2,3
    v2f  B1t[4];
    float bias1v[4], bias2v[4];
    #pragma unroll
    for (int nt = 0; nt < 4; ++nt) {
        const int colN = nt*16 + m;
        const int kx   = hiH ? 2 : 0;
        v2f b; b.x = sW1[kx*64 + colN]; b.y = sW1[(kx+1)*64 + colN];
        B1t[nt]   = b;
        bias1v[nt] = sb1[colN];
        bias2v[nt] = sb2[colN];
    }
    const float bias3v = sb3[m];

    __bf16* hb = sH[wave];
    float*  lg = sLg[wave];
    const int rowBase = hiH ? 8 : 0;     // D-matrix row group for this half-wave
    const int waveGlobal = blk*WAVES + wave;

    for (int it = 0; it < tilesPerWave; ++it) {
        const int tile = waveGlobal * tilesPerWave + it;
        const long long s    = (long long)tile*16 + m;
        const long long gidx = s * NCH + k;
        const float tg = tg_[gidx], tl = tl_[gidx], ti = ti_[gidx];
        const float mu = mu_[s],  mub = mub_[s];
        // prefetch next tile's scattered operands (speculative, safe at tail)
        __builtin_prefetch(&tg_[gidx + 16LL*NCH], 0, 1);
        __builtin_prefetch(&tl_[gidx + 16LL*NCH], 0, 1);
        __builtin_prefetch(&ti_[gidx + 16LL*NCH], 0, 1);
        const float tauTot = tg + tl + ti;
        const float t_dir = __expf(-tauTot / (mu  + 1e-5f));
        const float t_dif = __expf(-tauTot / (mub + 1e-3f));
        float e[2][3];

        #pragma unroll
        for (int p = 0; p < 2; ++p) {
            const float x3 = p ? mub : mu;

            // ---- layer 1: 4->64 via 4x v_wmma_f32_16x16x4_f32 ----
            // A (16x4 f32): lanes 0-15 -> {K0=tau_g, K1=tau_lw}; lanes 16-31 -> {K2=tau_iw, K3=mu}
            v2f a; a.x = hiH ? ti : tg; a.y = hiH ? x3 : tl;
            #pragma unroll
            for (int nt = 0; nt < 4; ++nt) {
                const float bv = bias1v[nt];
                v8f c = { bv, bv, bv, bv, bv, bv, bv, bv };
                c = __builtin_amdgcn_wmma_f32_16x16x4_f32(false, a, false, B1t[nt], (short)0, c, false, false);
                #pragma unroll
                for (int r = 0; r < 8; ++r)
                    ((unsigned short*)hb)[(rowBase + r)*64 + nt*16 + m] = f2bf(eluf(c[r]));
            }
            asm volatile("s_wait_dscnt 0x0" ::: "memory");

            // ---- A tiles (16x32 bf16 lane layout) from LDS ----
            v16bf A[2];
            #pragma unroll
            for (int kt = 0; kt < 2; ++kt) {
                const int kk = kt*32 + (hiH ? 16 : 0);
                BU au;
                au.u[0] = *(const uint4*)&hb[m*64 + kk];
                au.u[1] = *(const uint4*)&hb[m*64 + kk + 8];
                A[kt] = au.v;
            }

            // ---- layer 2: 64->64 via 8x v_wmma_f32_16x16x32_bf16 ----
            #pragma unroll
            for (int nt = 0; nt < 4; ++nt) {
                const float bv = bias2v[nt];
                v8f c = { bv, bv, bv, bv, bv, bv, bv, bv };
                c = __builtin_amdgcn_wmma_f32_16x16x32_bf16(false, A[0], false, B2t[0][nt], (short)0, c, false, false);
                c = __builtin_amdgcn_wmma_f32_16x16x32_bf16(false, A[1], false, B2t[1][nt], (short)0, c, false, false);
                #pragma unroll
                for (int r = 0; r < 8; ++r)
                    ((unsigned short*)hb)[(rowBase + r)*64 + nt*16 + m] = f2bf(eluf(c[r]));
            }
            asm volatile("s_wait_dscnt 0x0" ::: "memory");

            // ---- layer 3: 64->3 (N padded to 16) via 2 WMMAs ----
            v16bf A2[2];
            #pragma unroll
            for (int kt = 0; kt < 2; ++kt) {
                const int kk = kt*32 + (hiH ? 16 : 0);
                BU au;
                au.u[0] = *(const uint4*)&hb[m*64 + kk];
                au.u[1] = *(const uint4*)&hb[m*64 + kk + 8];
                A2[kt] = au.v;
            }
            v8f c3 = { bias3v, bias3v, bias3v, bias3v, bias3v, bias3v, bias3v, bias3v };
            c3 = __builtin_amdgcn_wmma_f32_16x16x32_bf16(false, A2[0], false, B3t[0], (short)0, c3, false, false);
            c3 = __builtin_amdgcn_wmma_f32_16x16x32_bf16(false, A2[1], false, B3t[1], (short)0, c3, false, false);

            if (m < 3) {     // column-major logits: lane stores its 8 rows contiguously
                float tmp[8];
                #pragma unroll
                for (int r = 0; r < 8; ++r) tmp[r] = c3[r];
                uint4* dst = (uint4*)&lg[m*16 + rowBase];
                dst[0] = *(uint4*)&tmp[0];
                dst[1] = *(uint4*)&tmp[4];
            }
            asm volatile("s_wait_dscnt 0x0" ::: "memory");

            // ---- softplus + 3-way softmax (duplicated across half-waves) ----
            {
                const float s0 = softplusf(lg[0*16 + m]);
                const float s1 = softplusf(lg[1*16 + m]);
                const float s2 = softplusf(lg[2*16 + m]);
                const float mx = fmaxf(s0, fmaxf(s1, s2));
                const float e0 = __expf(s0-mx), e1 = __expf(s1-mx), e2 = __expf(s2-mx);
                const float inv = 1.f / (e0+e1+e2);
                e[p][0] = e0*inv; e[p][1] = e1*inv; e[p][2] = e2*inv;
            }
        }

        if (lane < 16) {
            float4 o0, o1;
            o0.x = t_dir;   o0.y = t_dif;   o0.z = e[0][0]; o0.w = e[0][1];
            o1.x = e[0][2]; o1.y = e[1][0]; o1.z = e[1][1]; o1.w = e[1][2];
            float4* po = (float4*)&out[gidx*8];
            po[0] = o0; po[1] = o1;
        }
    }
}

extern "C" void kernel_launch(void* const* d_in, const int* in_sizes, int n_in,
                              void* d_out, int out_size, void* d_ws, size_t ws_size,
                              hipStream_t stream) {
    const float* tg  = (const float*)d_in[0];
    const float* tl  = (const float*)d_in[1];
    const float* ti  = (const float*)d_in[2];
    const float* mu  = (const float*)d_in[3];
    const float* mub = (const float*)d_in[4];
    const float* W1  = (const float*)d_in[5];
    const float* b1  = (const float*)d_in[6];
    const float* W2  = (const float*)d_in[7];
    const float* b2  = (const float*)d_in[8];
    const float* W3  = (const float*)d_in[9];
    const float* b3  = (const float*)d_in[10];
    float* out = (float*)d_out;

    const int B = in_sizes[3];                       // mu has B elements
    const int tilesPerChannel = B / 16;              // 16384 for B=262144
    const int tilesPerWave = tilesPerChannel / (NBLK_PER_CH * WAVES);

    dim3 grid(NCH * NBLK_PER_CH), block(256);
    lp_kernel<<<grid, block, 0, stream>>>(tg, tl, ti, mu, mub,
                                          W1, b1, W2, b2, W3, b3,
                                          out, tilesPerWave);
}
